// CausalSALayer_12266426597820
// MI455X (gfx1250) — matrix operand order repaired and checked by reading, not compiled, not run on hardware
//
#include <hip/hip_runtime.h>
#include <hip/hip_bf16.h>

// ---------------------------------------------------------------------------
// Types / WMMA helpers (CDNA5 / gfx1250, wave32)
// ---------------------------------------------------------------------------
typedef __attribute__((ext_vector_type(16))) _Float16 v16h;
typedef __attribute__((ext_vector_type(8)))  _Float16 h8;
typedef __attribute__((ext_vector_type(8)))  float    v8f;
typedef int v4i __attribute__((vector_size(16)));   // 4 x i32 (16B chunk)

#define AS1 __attribute__((address_space(1)))
#define AS3 __attribute__((address_space(3)))

// Async global->LDS copy engine (ASYNCcnt) if the toolchain exposes it.
#if __has_builtin(__builtin_amdgcn_global_load_async_to_lds_b128) && \
    __has_builtin(__builtin_amdgcn_s_wait_asynccnt)
#define USE_ASYNC 1
#define ASYNC_WAIT() __builtin_amdgcn_s_wait_asynccnt(0)
#else
#define USE_ASYNC 0
#define ASYNC_WAIT()
#endif

union H16 { v16h v; h8 h[2]; };

__device__ __forceinline__ v8f wmma32(v16h a, v16h b, v8f c) {
  // v_wmma_f32_16x16x32_f16: D = A(16x32) * B(32x16) + C(16x16 f32)
  return __builtin_amdgcn_wmma_f32_16x16x32_f16(
      /*neg_a=*/false, a, /*neg_b=*/false, b,
      /*c_mod=*/(short)0, c, /*reuse_a=*/false, /*reuse_b=*/false);
}

// A fragment (16x32 f16, MxK), source row-major [m][k], stride in halves.
// ISA 7.12.2: lanes 0-15 row m hold K {0..7,16..23}; lanes 16-31 hold {8..15,24..31}.
__device__ __forceinline__ v16h load_a_frag(const _Float16* base, int stride) {
  const int lane = threadIdx.x & 31;
  const int m = lane & 15, hi = lane >> 4;
  H16 u;
  u.h[0] = *(const h8*)(base + m * stride + hi * 8);
  u.h[1] = *(const h8*)(base + m * stride + 16 + hi * 8);
  return u.v;
}

// B fragment (32x16 f16, KxN), source row-major [n][k] (i.e. W rows), stride halves.
// Lane n (0-15): column n, K {0..7,16..23}; lane n+16: column n, K {8..15,24..31}.
__device__ __forceinline__ v16h load_b_frag(const _Float16* base, int stride) {
  const int lane = threadIdx.x & 31;
  const int n = lane & 15, hi = lane >> 4;
  H16 u;
  u.h[0] = *(const h8*)(base + n * stride + hi * 8);
  u.h[1] = *(const h8*)(base + n * stride + 16 + hi * 8);
  return u.v;
}

__device__ __forceinline__ float gelu_tanh(float x) {
  const float k0 = 0.7978845608028654f;   // sqrt(2/pi)
  const float k1 = 0.044715f;
  float t = tanhf(k0 * (x + k1 * x * x * x));
  return 0.5f * x * (1.0f + t);
}

// ---------------------------------------------------------------------------
// Model constants
// ---------------------------------------------------------------------------
#define DM     1024
#define NHEAD  16
#define HD     64
#define FFND   4096
#define BSZ    2
#define SEQ    2048
#define MROWS  (BSZ * SEQ)      // 4096 token rows

// ---------------------------------------------------------------------------
// f32 -> f16 cast (8 elems / thread, vectorized)
// ---------------------------------------------------------------------------
__global__ __launch_bounds__(256) void cast_f32_to_f16(
    const float* __restrict__ in, _Float16* __restrict__ out, int n8) {
  int i = blockIdx.x * blockDim.x + threadIdx.x;
  if (i >= n8) return;
  const float4* p = (const float4*)in;
  float4 a = p[2 * i], b = p[2 * i + 1];
  h8 o;
  o[0] = (_Float16)a.x; o[1] = (_Float16)a.y; o[2] = (_Float16)a.z; o[3] = (_Float16)a.w;
  o[4] = (_Float16)b.x; o[5] = (_Float16)b.y; o[6] = (_Float16)b.z; o[7] = (_Float16)b.w;
  ((h8*)out)[i] = o;
}

// ---------------------------------------------------------------------------
// Tiled GEMM: C[M,N] = A[M,K] * W[N,K]^T (+ epilogue)
//   EPI 0: out16 = C                  (QKV projections)
//   EPI 1: out32 = C + bias + resid   (Wo / FFN2 + residual add)
//   EPI 2: out16 = gelu(C + bias)     (FFN1)
// Block: 256 thr (8 waves, 4 M-waves x 2 N-waves); tile 128x128.
// K-slab of 64 double-buffered in LDS via async global->LDS copies:
// 16 WMMAs per wave between barriers, copies overlap compute.
// ---------------------------------------------------------------------------
#define TS 72   // LDS K-slab stride in halves (64 + 8 pad, keeps 16B alignment)

template <int EPI>
__global__ __launch_bounds__(256) void gemm_kernel(
    const _Float16* __restrict__ A, const _Float16* __restrict__ W,
    int M, int Ncols, int K,
    const float* __restrict__ bias, const float* __restrict__ resid,
    _Float16* __restrict__ out16, float* __restrict__ out32) {
  __shared__ _Float16 Asm[2][128 * TS];
  __shared__ _Float16 Wsm[2][128 * TS];

  const int tid  = threadIdx.x;
  const int wave = tid >> 5;
  const int wm   = wave & 3;        // 0..3  (M direction, 32 rows each)
  const int wn   = wave >> 2;       // 0..1  (N direction, 64 cols each)
  const int lane = tid & 31;
  const int hi   = lane >> 4, cn = lane & 15;

  const int rowA = blockIdx.y * 128;
  const int rowW = blockIdx.x * 128;

  // Stage a 128x64 f16 slab of A and of W into LDS buffer `buf`.
  auto stage = [&](int buf, int k0) {
#pragma unroll
    for (int i = 0; i < 4; ++i) {
      int idx = tid + i * 256;          // 1024 chunks of 8 halves (16B)
      int row = idx >> 3, c8 = idx & 7;
      const _Float16* ga = A + (size_t)(rowA + row) * K + k0 + c8 * 8;
      const _Float16* gw = W + (size_t)(rowW + row) * K + k0 + c8 * 8;
      _Float16* la = &Asm[buf][row * TS + c8 * 8];
      _Float16* lw = &Wsm[buf][row * TS + c8 * 8];
#if USE_ASYNC
      __builtin_amdgcn_global_load_async_to_lds_b128(
          (AS1 v4i*)ga, (AS3 v4i*)la, 0, 0);
      __builtin_amdgcn_global_load_async_to_lds_b128(
          (AS1 v4i*)gw, (AS3 v4i*)lw, 0, 0);
#else
      *(h8*)la = *(const h8*)ga;
      *(h8*)lw = *(const h8*)gw;
#endif
    }
  };

  v8f acc[2][4];
#pragma unroll
  for (int mi = 0; mi < 2; ++mi)
#pragma unroll
    for (int ni = 0; ni < 4; ++ni) acc[mi][ni] = (v8f)0.0f;

  stage(0, 0);
  ASYNC_WAIT();
  __syncthreads();

  const int nIter = K >> 6;             // K-slabs of 64
  for (int it = 0; it < nIter; ++it) {
    const int cur = it & 1;
    if (it + 1 < nIter) stage(cur ^ 1, (it + 1) << 6);

    const _Float16* Ab = Asm[cur];
    const _Float16* Wb = Wsm[cur];
#pragma unroll
    for (int ks = 0; ks < 64; ks += 32) {
      v16h a0 = load_a_frag(&Ab[(wm * 32 + 0)  * TS + ks], TS);
      v16h a1 = load_a_frag(&Ab[(wm * 32 + 16) * TS + ks], TS);
#pragma unroll
      for (int ni = 0; ni < 4; ++ni) {
        v16h b = load_b_frag(&Wb[(wn * 64 + ni * 16) * TS + ks], TS);
        acc[0][ni] = wmma32(a0, b, acc[0][ni]);
        acc[1][ni] = wmma32(a1, b, acc[1][ni]);
      }
    }
    ASYNC_WAIT();        // our writes into the other buffer are done
    __syncthreads();     // everyone done writing next / reading current
  }

  // Epilogue. C layout: VGPR r -> row (r + hi*8), col = cn within 16x16 frag.
#pragma unroll
  for (int mi = 0; mi < 2; ++mi) {
#pragma unroll
    for (int ni = 0; ni < 4; ++ni) {
      int col = rowW + wn * 64 + ni * 16 + cn;
#pragma unroll
      for (int r = 0; r < 8; ++r) {
        int row = rowA + wm * 32 + mi * 16 + r + hi * 8;
        float v = acc[mi][ni][r];
        size_t idx = (size_t)row * Ncols + col;
        if (EPI == 0) {
          out16[idx] = (_Float16)v;
        } else if (EPI == 1) {
          out32[idx] = v + bias[col] + resid[idx];
        } else {
          out16[idx] = (_Float16)gelu_tanh(v + bias[col]);
        }
      }
    }
  }
}

// ---------------------------------------------------------------------------
// LayerNorm: one block (256 thr) per 1024-wide row.
// Optionally writes an f16 copy for the next GEMM.
// ---------------------------------------------------------------------------
__global__ __launch_bounds__(256) void ln_kernel(
    const float* __restrict__ in, const float* __restrict__ g,
    const float* __restrict__ be, float* __restrict__ o32,
    _Float16* __restrict__ o16) {
  __shared__ float red[256];
  const int row = blockIdx.x, t = threadIdx.x;
  const float* x = in + (size_t)row * DM;

  float v[4], s = 0.0f;
#pragma unroll
  for (int i = 0; i < 4; ++i) { v[i] = x[t * 4 + i]; s += v[i]; }
  red[t] = s; __syncthreads();
  for (int off = 128; off > 0; off >>= 1) {
    if (t < off) red[t] += red[t + off];
    __syncthreads();
  }
  float mu = red[0] * (1.0f / DM);
  __syncthreads();

  float s2 = 0.0f;
#pragma unroll
  for (int i = 0; i < 4; ++i) { float d = v[i] - mu; s2 += d * d; }
  red[t] = s2; __syncthreads();
  for (int off = 128; off > 0; off >>= 1) {
    if (t < off) red[t] += red[t + off];
    __syncthreads();
  }
  float rstd = rsqrtf(red[0] * (1.0f / DM) + 1e-5f);

#pragma unroll
  for (int i = 0; i < 4; ++i) {
    int c = t * 4 + i;
    float y = (v[i] - mu) * rstd * g[c] + be[c];
    o32[(size_t)row * DM + c] = y;
    if (o16) o16[(size_t)row * DM + c] = (_Float16)y;
  }
}

// ---------------------------------------------------------------------------
// Causal flash attention.
// Grid: (SEQ/64, B*NHEAD). Block: 128 thr = 4 waves; wave w owns 16 query rows.
// Key/value processed in tiles of 32; V stored transposed in LDS so P*V uses
// the standard B-fragment loader. Online softmax with 16-lane shuffles.
// ---------------------------------------------------------------------------
#define KSTR 72   // K tile LDS stride (64 + 8)
#define VSTR 40   // V^T / P LDS stride (32 + 8)

__global__ __launch_bounds__(128) void attn_kernel(
    const _Float16* __restrict__ Q, const _Float16* __restrict__ Kt,
    const _Float16* __restrict__ V, _Float16* __restrict__ ctx) {
  __shared__ _Float16 Ksm[32 * KSTR];        // [j][d]   (keys)
  __shared__ _Float16 Vts[64 * VSTR];        // [d][j]   (values, transposed)
  __shared__ _Float16 Psm[4 * 16 * VSTR];    // per-wave [m][j] probs scratch

  const int tid  = threadIdx.x;
  const int wave = tid >> 5;
  const int lane = tid & 31;
  const int hi   = lane >> 4, cn = lane & 15;

  const int bh = blockIdx.y;
  const int b  = bh / NHEAD, h = bh % NHEAD;
  const int q0 = blockIdx.x * 64;
  const int qw = q0 + wave * 16;             // this wave's first query row

  // Q fragments for this wave's 16 rows (16x64 split into 2 K-chunks of 32).
  const _Float16* qbase = Q + ((size_t)(b * SEQ + qw)) * DM + h * HD;
  v16h qf0 = load_a_frag(qbase,      DM);
  v16h qf1 = load_a_frag(qbase + 32, DM);

  v8f acc[4];
#pragma unroll
  for (int ni = 0; ni < 4; ++ni) acc[ni] = (v8f)0.0f;
  float m_run[8], l_run[8];
#pragma unroll
  for (int r = 0; r < 8; ++r) { m_run[r] = -1e30f; l_run[r] = 0.0f; }

  _Float16* Pw = &Psm[wave * 16 * VSTR];

  for (int j0 = 0; j0 < q0 + 64; j0 += 32) {
    // --- stage K tile [32][64] and V tile transposed [64][32] ---
#pragma unroll
    for (int i = 0; i < 2; ++i) {
      int idx = tid + i * 128;               // 256 chunks of 8 halves
      int row = idx >> 3, c8 = idx & 7;
      size_t gsrc = ((size_t)(b * SEQ + j0 + row)) * DM + h * HD + c8 * 8;
      *(h8*)&Ksm[row * KSTR + c8 * 8] = *(const h8*)(Kt + gsrc);
      h8 vv = *(const h8*)(V + gsrc);
#pragma unroll
      for (int e = 0; e < 8; ++e) Vts[(c8 * 8 + e) * VSTR + row] = vv[e];
    }
    __syncthreads();

    if (j0 <= qw + 15) {                     // wave-uniform causal tile skip
      // --- S = Q * K^T (16x32), two 16x16 frags, K-dim = 64 in two steps ---
      v8f s0 = (v8f)0.0f, s1 = (v8f)0.0f;
      {
        v16h bk = load_b_frag(&Ksm[0 * KSTR + 0],  KSTR);
        s0 = wmma32(qf0, bk, s0);
        bk = load_b_frag(&Ksm[0 * KSTR + 32], KSTR);
        s0 = wmma32(qf1, bk, s0);
        bk = load_b_frag(&Ksm[16 * KSTR + 0],  KSTR);
        s1 = wmma32(qf0, bk, s1);
        bk = load_b_frag(&Ksm[16 * KSTR + 32], KSTR);
        s1 = wmma32(qf1, bk, s1);
      }

      // --- scale + causal mask + online softmax ---
      float sv0[8], sv1[8], mt[8];
#pragma unroll
      for (int r = 0; r < 8; ++r) {
        int qg = qw + r + hi * 8;
        int jg0 = j0 + cn, jg1 = j0 + 16 + cn;
        sv0[r] = (jg0 > qg) ? -1e30f : s0[r] * 0.125f;   // 1/sqrt(64)
        sv1[r] = (jg1 > qg) ? -1e30f : s1[r] * 0.125f;
        mt[r] = fmaxf(sv0[r], sv1[r]);
      }
#pragma unroll
      for (int off = 8; off >= 1; off >>= 1)
#pragma unroll
        for (int r = 0; r < 8; ++r)
          mt[r] = fmaxf(mt[r], __shfl_xor(mt[r], off, 16));

      float al[8], rs[8];
#pragma unroll
      for (int r = 0; r < 8; ++r) {
        float mn = fmaxf(m_run[r], mt[r]);
        al[r] = __expf(m_run[r] - mn);
        m_run[r] = mn;
        float p0 = __expf(sv0[r] - mn);
        float p1 = __expf(sv1[r] - mn);
        rs[r] = p0 + p1;
        Pw[(r + hi * 8) * VSTR + cn]      = (_Float16)p0;
        Pw[(r + hi * 8) * VSTR + 16 + cn] = (_Float16)p1;
      }
#pragma unroll
      for (int off = 8; off >= 1; off >>= 1)
#pragma unroll
        for (int r = 0; r < 8; ++r)
          rs[r] += __shfl_xor(rs[r], off, 16);
#pragma unroll
      for (int r = 0; r < 8; ++r) l_run[r] = l_run[r] * al[r] + rs[r];
#pragma unroll
      for (int ni = 0; ni < 4; ++ni)
#pragma unroll
        for (int r = 0; r < 8; ++r) acc[ni][r] *= al[r];

      // --- O += P(16x32) * V(32x64) ---
      v16h pf = load_a_frag(Pw, VSTR);
#pragma unroll
      for (int ni = 0; ni < 4; ++ni) {
        v16h bv = load_b_frag(&Vts[(ni * 16) * VSTR], VSTR);
        acc[ni] = wmma32(pf, bv, acc[ni]);
      }
    }
    __syncthreads();
  }

  // --- normalize + write context [b, q, h*64 + d] as f16 ---
#pragma unroll
  for (int ni = 0; ni < 4; ++ni)
#pragma unroll
    for (int r = 0; r < 8; ++r) {
      float o = acc[ni][r] / l_run[r];
      size_t dst = ((size_t)(b * SEQ + qw + r + hi * 8)) * DM
                 + h * HD + ni * 16 + cn;
      ctx[dst] = (_Float16)o;
    }
}

// ---------------------------------------------------------------------------
// Host orchestration
// ---------------------------------------------------------------------------
extern "C" void kernel_launch(void* const* d_in, const int* in_sizes, int n_in,
                              void* d_out, int out_size, void* d_ws, size_t ws_size,
                              hipStream_t stream) {
  (void)in_sizes; (void)n_in; (void)out_size; (void)ws_size;

  const float* tgt   = (const float*)d_in[0];
  // d_in[1]=tgt_mask (causal, reimplemented arithmetically), d_in[2]=pad (all false)
  const float* Wq    = (const float*)d_in[3];
  const float* Wk    = (const float*)d_in[4];
  const float* Wv    = (const float*)d_in[5];
  const float* Wo    = (const float*)d_in[6];
  const float* bo    = (const float*)d_in[7];
  const float* W1    = (const float*)d_in[8];
  const float* b1    = (const float*)d_in[9];
  const float* W2    = (const float*)d_in[10];
  const float* b2    = (const float*)d_in[11];
  const float* g1    = (const float*)d_in[12];
  const float* be1   = (const float*)d_in[13];
  const float* g2    = (const float*)d_in[14];
  const float* be2   = (const float*)d_in[15];
  float* out = (float*)d_out;

  size_t off = 0;
  auto alloc = [&](size_t bytes) -> void* {
    void* p = (char*)d_ws + off;
    off += (bytes + 255) & ~(size_t)255;
    return p;
  };
  _Float16* X16   = (_Float16*)alloc((size_t)MROWS * DM * 2);
  _Float16* Wq16  = (_Float16*)alloc((size_t)DM * DM * 2);
  _Float16* Wk16  = (_Float16*)alloc((size_t)DM * DM * 2);
  _Float16* Wv16  = (_Float16*)alloc((size_t)DM * DM * 2);
  _Float16* Wo16  = (_Float16*)alloc((size_t)DM * DM * 2);
  _Float16* W116  = (_Float16*)alloc((size_t)FFND * DM * 2);
  _Float16* W216  = (_Float16*)alloc((size_t)DM * FFND * 2);
  _Float16* Q16   = (_Float16*)alloc((size_t)MROWS * DM * 2);
  _Float16* K16   = (_Float16*)alloc((size_t)MROWS * DM * 2);
  _Float16* V16   = (_Float16*)alloc((size_t)MROWS * DM * 2);
  _Float16* CTX16 = (_Float16*)alloc((size_t)MROWS * DM * 2);
  float*    RES1  = (float*)   alloc((size_t)MROWS * DM * 4);
  float*    X1f   = (float*)   alloc((size_t)MROWS * DM * 4);
  _Float16* X1h   = (_Float16*)alloc((size_t)MROWS * DM * 2);
  _Float16* H1    = (_Float16*)alloc((size_t)MROWS * FFND * 2);
  float*    RES2  = (float*)   alloc((size_t)MROWS * DM * 4);

  auto cast = [&](const float* src, _Float16* dst, size_t n) {
    int n8 = (int)(n / 8);
    cast_f32_to_f16<<<(n8 + 255) / 256, 256, 0, stream>>>(src, dst, n8);
  };
  cast(tgt, X16,  (size_t)MROWS * DM);
  cast(Wq,  Wq16, (size_t)DM * DM);
  cast(Wk,  Wk16, (size_t)DM * DM);
  cast(Wv,  Wv16, (size_t)DM * DM);
  cast(Wo,  Wo16, (size_t)DM * DM);
  cast(W1,  W116, (size_t)FFND * DM);
  cast(W2,  W216, (size_t)DM * FFND);

  // QKV projections (f16 out)
  dim3 gQKV(DM / 128, MROWS / 128);
  gemm_kernel<0><<<gQKV, 256, 0, stream>>>(X16, Wq16, MROWS, DM, DM,
                                           nullptr, nullptr, Q16, nullptr);
  gemm_kernel<0><<<gQKV, 256, 0, stream>>>(X16, Wk16, MROWS, DM, DM,
                                           nullptr, nullptr, K16, nullptr);
  gemm_kernel<0><<<gQKV, 256, 0, stream>>>(X16, Wv16, MROWS, DM, DM,
                                           nullptr, nullptr, V16, nullptr);

  // Causal flash attention -> context (f16)
  attn_kernel<<<dim3(SEQ / 64, BSZ * NHEAD), 128, 0, stream>>>(Q16, K16, V16, CTX16);

  // h = ctx @ Wo^T + bo; RES1 = tgt + h   (f32)
  gemm_kernel<1><<<gQKV, 256, 0, stream>>>(CTX16, Wo16, MROWS, DM, DM,
                                           bo, tgt, nullptr, RES1);
  // x1 = LN(RES1) -> f32 + f16 copies
  ln_kernel<<<MROWS, 256, 0, stream>>>(RES1, g1, be1, X1f, X1h);

  // H1 = gelu(x1 @ W1^T + b1)   (f16, 4096 wide)
  dim3 gF1(FFND / 128, MROWS / 128);
  gemm_kernel<2><<<gF1, 256, 0, stream>>>(X1h, W116, MROWS, FFND, DM,
                                          b1, nullptr, H1, nullptr);
  // RES2 = x1 + (H1 @ W2^T + b2)   (f32)
  gemm_kernel<1><<<gQKV, 256, 0, stream>>>(H1, W216, MROWS, DM, FFND,
                                           b2, X1f, nullptr, RES2);
  // out = LN(RES2)
  ln_kernel<<<MROWS, 256, 0, stream>>>(RES2, g2, be2, out, nullptr);
}
